// a3d_module_22617297781399
// MI455X (gfx1250) — matrix operand order repaired and checked by reading, not compiled, not run on hardware
//
#include <hip/hip_runtime.h>
#include <hip/hip_bf16.h>

// ---------------------------------------------------------------------------
// Problem constants (B,T,H,W,C = 4,16,16,16,512 ; RC = 128)
// ---------------------------------------------------------------------------
#define BATCH 4
#define NTOK  4096          // T*H*W
#define CCH   512
#define RCH   128
#define ROWS  16384         // BATCH*NTOK
#define SOFTMAX_SCALE 0.0027621359f   // 1/sqrt(16*16*512)

typedef __attribute__((ext_vector_type(16))) __bf16 v16bf;
typedef __attribute__((ext_vector_type(2)))  __bf16 v2bf;
typedef __attribute__((ext_vector_type(8)))  float  v8f;

// ---------------------------------------------------------------------------
// WMMA wrapper: D(f32 16x16) = A(bf16 16x32) * B(bf16 32x16) + C
// ---------------------------------------------------------------------------
static __device__ __forceinline__ v8f wmma_bf16(v16bf a, v16bf b, v8f c) {
  return __builtin_amdgcn_wmma_f32_16x16x32_bf16(
      /*neg_a=*/false, a, /*neg_b=*/false, b,
      /*c_mod=*/(short)0, c, /*reuse_a=*/false, /*reuse_b=*/false);
}

// Load a full 16 x bf16 fragment as two 16B LDS/global reads (ds_load_b128 x2)
static __device__ __forceinline__ v16bf frag_b128x2(const __bf16* p) {
  union { uint4 u[2]; v16bf v; } c;
  const uint4* q = (const uint4*)p;
  c.u[0] = q[0];
  c.u[1] = q[1];
  return c.v;
}

// ---------------------------------------------------------------------------
// Fragment loaders (wave32), contiguous k order shared by A and B:
//  A (16x32, 16-bit): lane l<16 -> M=l, K=0..15; lane>=16 -> K=16..31
//  B (32x16, 16-bit): lane l<16 -> N=l, K=0..15; lane>=16 -> K=16..31
//  C/D (16x16 f32):   elem r -> M = r + (lane>=16 ? 8 : 0), N = lane%16
// ---------------------------------------------------------------------------
static __device__ __forceinline__ v16bf a_frag_f32(const float* src, int ld,
                                                   int Mb, int kb, int l16, int hi) {
  const float* row = src + (size_t)(Mb + l16) * ld + kb + hi * 16;
  v16bf a;
#pragma unroll
  for (int e = 0; e < 8; ++e) a[e] = (__bf16)row[e];
#pragma unroll
  for (int e = 0; e < 8; ++e) a[8 + e] = (__bf16)row[8 + e];
  return a;
}

static __device__ __forceinline__ v16bf a_frag_bf16(const __bf16* src, int ld,
                                                    int Mb, int kb, int l16, int hi) {
  // two contiguous 8 x bf16 runs -> two 16B loads
  const __bf16* row = src + (size_t)(Mb + l16) * ld + kb + hi * 16;
  union { uint4 u[2]; v16bf v; } c;
  c.u[0] = *(const uint4*)(row);
  c.u[1] = *(const uint4*)(row + 8);
  return c.v;
}

// ---------------------------------------------------------------------------
// Kernel 1: k/q/v pointwise projections (1x1x1 conv == GEMM).
// Weight tile pre-swizzled into WMMA B-fragment order in LDS:
//   sW[(chunk*32 + lane)*16 + e] = bf16( W[chunk*32 + (lane/16)*16 + e][Nb + lane%16] )
// so each lane's fragment is 32 contiguous bytes (2 x ds_load_b128).
// Grid: 3 proj * 8 N-tiles * 128 M-blocks = 3072 blocks (8 waves each).
// ---------------------------------------------------------------------------
__global__ __launch_bounds__(256) void qkv_proj_kernel(
    const float* __restrict__ x,
    const float* __restrict__ kw, const float* __restrict__ kbias,
    const float* __restrict__ qw, const float* __restrict__ qbias,
    const float* __restrict__ vw, const float* __restrict__ vbias,
    __bf16* __restrict__ kout, __bf16* __restrict__ qout, __bf16* __restrict__ vout) {
  __shared__ __align__(16) __bf16 sW[(CCH / 32) * 32 * 16];   // 16 KB

  const int lane = threadIdx.x & 31;
  const int wave = threadIdx.x >> 5;
  const int l16 = lane & 15, hi = (lane >> 4) & 1;

  const int proj = blockIdx.x >> 10;              // 1024 blocks per projection
  const int rem  = blockIdx.x & 1023;
  const int Nb   = (rem >> 7) << 4;               // 8 N-tiles of 16
  const int Mb   = ((rem & 127) << 7) + wave * 16;// 128 M-blocks of 128 rows

  const float* W  = (proj == 0) ? kw : ((proj == 1) ? qw : vw);
  const float* Bs = (proj == 0) ? kbias : ((proj == 1) ? qbias : vbias);
  __bf16* O       = (proj == 0) ? kout : ((proj == 1) ? qout : vout);

  // Stage weights: pack k-pairs -> one 4B LDS store each (e is even)
  for (int p = threadIdx.x; p < (CCH / 2) * 16; p += 256) {
    const int k = (p >> 4) * 2;
    const int n = p & 15;
    const float w0 = W[(size_t)k * RCH + Nb + n];
    const float w1 = W[(size_t)(k + 1) * RCH + Nb + n];
    const int ch = k >> 5;
    const int L  = ((k >> 4) & 1) * 16 + n;
    const int e  = k & 15;
    *(v2bf*)(sW + (ch * 32 + L) * 16 + e) = (v2bf){(__bf16)w0, (__bf16)w1};
  }
  __syncthreads();

  v8f acc = (v8f){};
#pragma unroll 4
  for (int kc = 0; kc < CCH; kc += 32) {
    v16bf a = a_frag_f32(x, CCH, Mb, kc, l16, hi);
    v16bf b = frag_b128x2(sW + ((kc >> 5) * 32 + lane) * 16);
    acc = wmma_bf16(a, b, acc);
  }
  const float bias = Bs[Nb + l16];
#pragma unroll
  for (int r = 0; r < 8; ++r)
    O[(size_t)(Mb + r + hi * 8) * RCH + Nb + l16] = (__bf16)(acc[r] + bias);
}

// ---------------------------------------------------------------------------
// Kernel 2: fused attention (flash-style, never materializes the 268MB score
// matrix). One block = 128 "i" rows (8 waves x 16 rows); loop over j in tiles
// of 32. Q staged row-major (score fragments are contiguous); V staged
// TRANSPOSED (sVt[c][j], software DS_LOAD_TR) so P@V fragments are contiguous.
// ---------------------------------------------------------------------------
__global__ __launch_bounds__(256) void flash_attn_kernel(
    const __bf16* __restrict__ kmat, const __bf16* __restrict__ qmat,
    const __bf16* __restrict__ vmat, float* __restrict__ attn) {
  __shared__ __align__(16) __bf16 sQ[32 * RCH];    // 8 KB, row-major [j][c]
  __shared__ __align__(16) __bf16 sVt[RCH * 32];   // 8 KB, transposed [c][j]
  __shared__ __align__(16) float sP[8][16 * 32];   // per-wave P transpose scratch

  const int lane = threadIdx.x & 31;
  const int wave = threadIdx.x >> 5;
  const int l16 = lane & 15, hi = (lane >> 4) & 1;
  const int bidx = blockIdx.x >> 5;                 // batch (32 i-blocks each)
  const int iblk = blockIdx.x & 31;
  const int ib = iblk * 128 + wave * 16;            // row base within batch

  const __bf16* kp = kmat + (size_t)bidx * NTOK * RCH;
  const __bf16* qp = qmat + (size_t)bidx * NTOK * RCH;
  const __bf16* vp = vmat + (size_t)bidx * NTOK * RCH;

  // Preload the wave's K rows as A fragments (full RC=128 -> 4 frags, 32 VGPRs)
  v16bf ka[4];
#pragma unroll
  for (int cc = 0; cc < 4; ++cc) ka[cc] = a_frag_bf16(kp, RCH, ib, cc * 32, l16, hi);

  v8f o[8];
#pragma unroll
  for (int n = 0; n < 8; ++n) o[n] = (v8f){};
  float m[8], l[8];
#pragma unroll
  for (int r = 0; r < 8; ++r) { m[r] = -3.0e38f; l[r] = 0.0f; }

  for (int jt = 0; jt < NTOK; jt += 32) {
    __syncthreads();  // previous iteration done consuming sQ/sVt
    {
      // Q: straight 8 KB copy (coalesced uint4)
      const uint4* gq = (const uint4*)(qp + (size_t)jt * RCH);
      uint4* sq = (uint4*)sQ;
      // V: coalesced uint4 read, transposed bf16 scatter into sVt[c][j]
      const uint4* gv = (const uint4*)(vp + (size_t)jt * RCH);
      for (int t = threadIdx.x; t < 512; t += 256) {
        sq[t] = gq[t];
        const uint4 d = gv[t];
        const int j  = t >> 4;            // source row in tile
        const int c0 = (t & 15) * 8;      // first of 8 source cols
        const __bf16* e8 = (const __bf16*)&d;
#pragma unroll
        for (int c = 0; c < 8; ++c) sVt[(c0 + c) * 32 + j] = e8[c];
      }
      if (jt + 32 < NTOK) {   // prefetch next tiles -> global_prefetch_b8
        __builtin_prefetch(qp + (size_t)(jt + 32) * RCH + threadIdx.x * 16, 0, 1);
        __builtin_prefetch(vp + (size_t)(jt + 32) * RCH + threadIdx.x * 16, 0, 1);
      }
    }
    __syncthreads();

    // S(16x32) = K_rows * Q_tile^T : two 16x16 N-subtiles, 4 K-chunks each.
    // B[k][n] = sQ[(nb+n)*RCH + kb+k]: lane fragment = 32 contiguous bytes.
    v8f s0 = (v8f){}, s1 = (v8f){};
#pragma unroll
    for (int cc = 0; cc < 4; ++cc) {
      v16bf b0 = frag_b128x2(sQ + (0 + l16) * RCH + cc * 32 + hi * 16);
      s0 = wmma_bf16(ka[cc], b0, s0);
      v16bf b1 = frag_b128x2(sQ + (16 + l16) * RCH + cc * 32 + hi * 16);
      s1 = wmma_bf16(ka[cc], b1, s1);
    }

    // Online softmax. C-layout row r lives in one 16-lane half; xor masks
    // 1,2,4,8 reduce within the half (bit 4 untouched) on wave32.
    float p0[8], p1[8];
#pragma unroll
    for (int r = 0; r < 8; ++r) {
      float a0 = s0[r] * SOFTMAX_SCALE;
      float a1 = s1[r] * SOFTMAX_SCALE;
      float mx = fmaxf(a0, a1);
      mx = fmaxf(mx, __shfl_xor(mx, 1, 32));
      mx = fmaxf(mx, __shfl_xor(mx, 2, 32));
      mx = fmaxf(mx, __shfl_xor(mx, 4, 32));
      mx = fmaxf(mx, __shfl_xor(mx, 8, 32));
      const float mn = fmaxf(m[r], mx);
      const float e0 = __expf(a0 - mn);
      const float e1 = __expf(a1 - mn);
      float rs = e0 + e1;
      rs += __shfl_xor(rs, 1, 32);
      rs += __shfl_xor(rs, 2, 32);
      rs += __shfl_xor(rs, 4, 32);
      rs += __shfl_xor(rs, 8, 32);
      const float co = __expf(m[r] - mn);
      l[r] = l[r] * co + rs;
      m[r] = mn;
      p0[r] = e0;
      p1[r] = e1;
#pragma unroll
      for (int n = 0; n < 8; ++n) o[n][r] *= co;
    }

    // Re-layout P (C/D layout f32) -> A-fragment bf16 via per-wave LDS scratch
    float* pw = &sP[wave][0];
#pragma unroll
    for (int r = 0; r < 8; ++r) {
      pw[(r + hi * 8) * 32 + l16]      = p0[r];
      pw[(r + hi * 8) * 32 + 16 + l16] = p1[r];
    }
    asm volatile("s_wait_dscnt 0x0" ::: "memory");   // wave-local DS RAW fence
    // A-frag elements e: j = hi*16+e -> one contiguous 16-float run,
    // read as float4 pairs (ds_load_b128).
    v16bf pa;
    {
      const float4* r0 = (const float4*)(pw + l16 * 32 + hi * 16);
      const float4* r1 = (const float4*)(pw + l16 * 32 + hi * 16 + 8);
      const float4 f0 = r0[0], f1 = r0[1], f2 = r1[0], f3 = r1[1];
      const float fs[16] = {f0.x, f0.y, f0.z, f0.w, f1.x, f1.y, f1.z, f1.w,
                            f2.x, f2.y, f2.z, f2.w, f3.x, f3.y, f3.z, f3.w};
#pragma unroll
      for (int e = 0; e < 16; ++e) pa[e] = (__bf16)fs[e];
    }

    // O(16x128) += P(16x32) @ V_tile(32x128)
    // B[k=j][n=c] = sVt[(nb+n)*32 + j]: lane fragment = 32 contiguous bytes.
#pragma unroll
    for (int n = 0; n < 8; ++n) {
      v16bf bv = frag_b128x2(sVt + (n * 16 + l16) * 32 + hi * 16);
      o[n] = wmma_bf16(pa, bv, o[n]);
    }
  }

  // Normalize by softmax denominator and store fp32 attention output
#pragma unroll
  for (int r = 0; r < 8; ++r) {
    const float inv = 1.0f / l[r];
    const size_t row = (size_t)bidx * NTOK + ib + r + hi * 8;
#pragma unroll
    for (int n = 0; n < 8; ++n)
      attn[row * RCH + n * 16 + l16] = o[n][r] * inv;
  }
}

// ---------------------------------------------------------------------------
// Kernel 3: restore projection  out = attn(16384x128) @ r_w(128x512) + r_b
// Same pre-swizzled weight-fragment LDS layout as kernel 1 (4 KB tile).
// Grid: 32 N-tiles * 128 M-blocks = 4096 blocks.
// ---------------------------------------------------------------------------
__global__ __launch_bounds__(256) void restore_proj_kernel(
    const float* __restrict__ attn, const float* __restrict__ rw,
    const float* __restrict__ rb, float* __restrict__ out) {
  __shared__ __align__(16) __bf16 sW[(RCH / 32) * 32 * 16];   // 4 KB

  const int lane = threadIdx.x & 31;
  const int wave = threadIdx.x >> 5;
  const int l16 = lane & 15, hi = (lane >> 4) & 1;

  const int Nb = (blockIdx.x >> 7) << 4;              // 32 N-tiles of 16
  const int Mb = ((blockIdx.x & 127) << 7) + wave * 16;

  for (int p = threadIdx.x; p < (RCH / 2) * 16; p += 256) {
    const int k = (p >> 4) * 2;
    const int n = p & 15;
    const float w0 = rw[(size_t)k * CCH + Nb + n];
    const float w1 = rw[(size_t)(k + 1) * CCH + Nb + n];
    const int ch = k >> 5;
    const int L  = ((k >> 4) & 1) * 16 + n;
    const int e  = k & 15;
    *(v2bf*)(sW + (ch * 32 + L) * 16 + e) = (v2bf){(__bf16)w0, (__bf16)w1};
  }
  __syncthreads();

  v8f acc = (v8f){};
#pragma unroll
  for (int kc = 0; kc < RCH; kc += 32) {
    v16bf a = a_frag_f32(attn, RCH, Mb, kc, l16, hi);
    v16bf b = frag_b128x2(sW + ((kc >> 5) * 32 + lane) * 16);
    acc = wmma_bf16(a, b, acc);
  }
  const float bias = rb[Nb + l16];
#pragma unroll
  for (int r = 0; r < 8; ++r)
    out[(size_t)(Mb + r + hi * 8) * CCH + Nb + l16] = acc[r] + bias;
}

// ---------------------------------------------------------------------------
// Host launcher. Workspace layout:
//   [0)                : k bf16   16384*128
//   [+16384*128)       : q bf16   16384*128
//   [+2*16384*128)     : v bf16   16384*128
//   [+3*16384*128 bf16): attn f32 16384*128       (total ~21 MB)
// ---------------------------------------------------------------------------
extern "C" void kernel_launch(void* const* d_in, const int* in_sizes, int n_in,
                              void* d_out, int out_size, void* d_ws, size_t ws_size,
                              hipStream_t stream) {
  (void)in_sizes; (void)n_in; (void)out_size; (void)ws_size;
  const float* x  = (const float*)d_in[0];
  const float* kw = (const float*)d_in[1];
  const float* kb = (const float*)d_in[2];
  const float* qw = (const float*)d_in[3];
  const float* qb = (const float*)d_in[4];
  const float* vw = (const float*)d_in[5];
  const float* vb = (const float*)d_in[6];
  const float* rw = (const float*)d_in[7];
  const float* rb = (const float*)d_in[8];
  float* out = (float*)d_out;

  __bf16* kbf = (__bf16*)d_ws;
  __bf16* qbf = kbf + (size_t)ROWS * RCH;
  __bf16* vbf = qbf + (size_t)ROWS * RCH;
  float*  att = (float*)(vbf + (size_t)ROWS * RCH);

  // 3 projections * 8 N-tiles * 128 M-blocks
  qkv_proj_kernel<<<3072, 256, 0, stream>>>(x, kw, kb, qw, qb, vw, vb,
                                            kbf, qbf, vbf);
  // 4 batches * 32 row-blocks
  flash_attn_kernel<<<128, 256, 0, stream>>>(kbf, qbf, vbf, att);
  // 32 N-tiles * 128 M-blocks
  restore_proj_kernel<<<4096, 256, 0, stream>>>(att, rw, rb, out);
}
